// AttentionBlock_74277164417687
// MI455X (gfx1250) — matrix operand order repaired and checked by reading, not compiled, not run on hardware
//
#include <hip/hip_runtime.h>
#include <hip/hip_bf16.h>
#include <stdint.h>

// ---------------------------------------------------------------------------
// CLIP-style transformer block on gfx1250 (MI455X).
// f16 WMMA (v_wmma_f32_16x16x32_f16) for all GEMMs + flash attention.
// Tile staging via Tensor Data Mover (tensor_load_to_lds + s_wait_tensorcnt),
// double-buffered in the GEMMs.
// ---------------------------------------------------------------------------

typedef _Float16 f16;
typedef __attribute__((ext_vector_type(16))) _Float16 v16h;
typedef __attribute__((ext_vector_type(8)))  float    v8f;
typedef __attribute__((ext_vector_type(4)))  unsigned int v4u;
typedef __attribute__((ext_vector_type(4)))  int      v4i;
typedef __attribute__((ext_vector_type(8)))  int      v8i;

#define D_MODEL 768
#define D_QKV   2304
#define D_FF    3072
#define SEQ_L   2048
#define BATCH   4
#define N_TOK   (SEQ_L * BATCH)   // 8192 tokens, row = l*B + b (x is [L,B,D])
#define NHEAD   12
#define HDIM    64

// ---- CDNA5 WMMA fragment swizzles (ISA 7.12.2, 16-bit A 16x32 / B 32x16) ---
__device__ __forceinline__ int kbaseA(int v, int hs) {
  return (v < 4) ? (hs * 8 + 2 * v) : (16 + hs * 8 + 2 * (v - 4));
}
__device__ __forceinline__ int kbaseB(int v, int hs) {
  return (v < 4) ? (hs * 8 + 2 * v) : (16 + hs * 8 + 2 * (v - 4));
}

// ---------------------------------------------------------------------------
// Tensor Data Mover: 2D tile (f16 elements) global -> LDS with row padding.
// D# per CDNA5 ISA ch.8: group0 = {count, lds_addr, global_addr, type=2},
// group1 = {data_size=2B, pad_enable, pad codes, tensor dims, tile dims,
// dim0 stride}. Groups 2/3 zero (<=2D tensor). Issued per-wave; completion
// tracked on TENSORcnt (in-order per wave).
// ---------------------------------------------------------------------------
__device__ __forceinline__ void tdm_load_2d_f16(
    uint32_t lds_addr, uint64_t gaddr, int tileW, int tileH,
    int rowStrideElems, int tdim0, int tdim1, int padIntCode, int padAmtCode) {
  v4u g0;
  g0.x = 1u;                                   // count=1 (valid), user mode
  g0.y = lds_addr;                             // LDS byte address
  g0.z = (uint32_t)gaddr;                      // global_addr[31:0]
  g0.w = (uint32_t)((gaddr >> 32) & 0x01FFFFFFu) | (2u << 30);  // type=2
  v8i g1;
  g1[0] = (1 << 16)                            // data_size: 2 bytes
        | (1 << 20)                            // pad_enable
        | (padIntCode << 22)                   // DWORDs between pads
        | (padAmtCode << 25);                  // pad DWORDs
  g1[1] = (tdim0 & 0xFFFF) << 16;              // tensor_dim0[15:0] @ bits 63:48
  g1[2] = (int)(((unsigned)tdim0 >> 16) | ((unsigned)(tdim1 & 0xFFFF) << 16));
  g1[3] = (int)(((unsigned)tdim1 >> 16) | ((unsigned)tileW << 16));  // tile_dim0
  g1[4] = tileH & 0xFFFF;                      // tile_dim1 (tile_dim2 = 0)
  g1[5] = rowStrideElems;                      // tensor_dim0_stride[31:0]
  g1[6] = 0;                                   // stride hi, dim1 stride lo
  g1[7] = 0;
  v4i gz = {0, 0, 0, 0};
#if __clang_major__ >= 23
  v8i gz8 = {0, 0, 0, 0, 0, 0, 0, 0};
  __builtin_amdgcn_tensor_load_to_lds(g0, g1, gz, gz, gz8, 0);
#else
  __builtin_amdgcn_tensor_load_to_lds(g0, g1, gz, gz, 0);
#endif
}

__device__ __forceinline__ uint32_t lds_off(const void* p) {
  return (uint32_t)(uintptr_t)p;   // LDS aperture: addr[31:0] = LDS offset
}

// ---------------------------------------------------------------------------
// fp32 -> f16 conversion (weights; deterministic, every launch)
// ---------------------------------------------------------------------------
__global__ __launch_bounds__(256) void f32_to_f16_kernel(
    const float* __restrict__ src, f16* __restrict__ dst, int n) {
  int i = blockIdx.x * blockDim.x + threadIdx.x;
  if (i < n) dst[i] = (f16)src[i];
}

// ---------------------------------------------------------------------------
// LayerNorm: one block per token row (D=768), 256 threads, 3 elems/thread.
// ---------------------------------------------------------------------------
__device__ __forceinline__ float wave_red_sum(float v) {
  #pragma unroll
  for (int m = 16; m >= 1; m >>= 1) v += __shfl_xor(v, m, 32);
  return v;
}

__global__ __launch_bounds__(256) void layernorm_kernel(
    const float* __restrict__ x, const float* __restrict__ g,
    const float* __restrict__ be, f16* __restrict__ out) {
  const int t = blockIdx.x;
  const int tid = threadIdx.x;
  const int lane = tid & 31, wave = tid >> 5;
  const float* row = x + (size_t)t * D_MODEL;
  float v0 = row[tid], v1 = row[tid + 256], v2 = row[tid + 512];
  float s = v0 + v1 + v2;
  float q = v0 * v0 + v1 * v1 + v2 * v2;
  __shared__ float rs[8], rq[8];
  s = wave_red_sum(s);
  q = wave_red_sum(q);
  if (lane == 0) { rs[wave] = s; rq[wave] = q; }
  __syncthreads();
  if (tid == 0) {
    float ts = 0.f, tq = 0.f;
    #pragma unroll
    for (int i = 0; i < 8; ++i) { ts += rs[i]; tq += rq[i]; }
    rs[0] = ts; rq[0] = tq;
  }
  __syncthreads();
  const float mean = rs[0] * (1.0f / D_MODEL);
  const float var  = rq[0] * (1.0f / D_MODEL) - mean * mean;
  const float rstd = rsqrtf(var + 1e-5f);
  f16* orow = out + (size_t)t * D_MODEL;
  orow[tid]       = (f16)((v0 - mean) * rstd * g[tid]       + be[tid]);
  orow[tid + 256] = (f16)((v1 - mean) * rstd * g[tid + 256] + be[tid + 256]);
  orow[tid + 512] = (f16)((v2 - mean) * rstd * g[tid + 512] + be[tid + 512]);
}

// ---------------------------------------------------------------------------
// WMMA GEMM:  C[M,N] = A[M,K](f16,row) * W[N,K](f16,row, torch Linear) + bias
// Block tile 64x64, BK=32, 256 threads = 8 waves (4 M-waves x 2 N-waves).
// Tiles staged by TDM (wave 0 issues), double-buffered on TENSORcnt.
// EPI 0: f16 out;  1: QuickGELU -> f16 out;  2: +fp32 residual -> fp32 out.
// ---------------------------------------------------------------------------
#define BM 64
#define BN 64
#define BK 32
#define LSTR 36   // LDS row stride in halves: 32 + 4 pad (TDM pad 16dw/2dw)

template <int EPI>
__global__ __launch_bounds__(256) void gemm_kernel(
    const f16* __restrict__ A, const f16* __restrict__ W,
    const float* __restrict__ bias, int M, int N, int K,
    f16* __restrict__ outH, float* __restrict__ outF,
    const float* __restrict__ resid) {
  __shared__ f16 ldsA[2][BM * LSTR];
  __shared__ f16 ldsB[2][BN * LSTR];
  const int tid  = threadIdx.x;
  const int lane = tid & 31, wave = tid >> 5;
  const int wm = wave & 3, wn = wave >> 2;
  const int bm0 = blockIdx.x * BM;
  const int bn0 = blockIdx.y * BN;
  const int r = lane & 15, hs = lane >> 4;

  v8f acc[2];
  #pragma unroll
  for (int nt = 0; nt < 2; ++nt) {
    const float bv = bias[bn0 + wn * 32 + nt * 16 + r];
    #pragma unroll
    for (int i = 0; i < 8; ++i) acc[nt][i] = bv;
  }

  const int nk = K / BK;
  // prologue: DMA tile 0 into buffer 0 (A + W: 2 tensor ops outstanding)
  if (wave == 0) {
    tdm_load_2d_f16(lds_off(&ldsA[0][0]),
                    (uint64_t)(uintptr_t)(A + (size_t)bm0 * K), BK, BM,
                    K, K, M, /*pad 16 dwords*/ 3, /*by 2 dwords*/ 1);
    tdm_load_2d_f16(lds_off(&ldsB[0][0]),
                    (uint64_t)(uintptr_t)(W + (size_t)bn0 * K), BK, BN,
                    K, K, N, 3, 1);
  }

  for (int kt = 0; kt < nk; ++kt) {
    const int buf = kt & 1;
    if (wave == 0) {
      if (kt + 1 < nk) {  // prefetch next tile into other buffer, then wait
        const int k1 = (kt + 1) * BK;
        tdm_load_2d_f16(lds_off(&ldsA[buf ^ 1][0]),
                        (uint64_t)(uintptr_t)(A + (size_t)bm0 * K + k1), BK, BM,
                        K, K, M, 3, 1);
        tdm_load_2d_f16(lds_off(&ldsB[buf ^ 1][0]),
                        (uint64_t)(uintptr_t)(W + (size_t)bn0 * K + k1), BK, BN,
                        K, K, N, 3, 1);
        __builtin_amdgcn_s_wait_tensorcnt(2);  // tile kt done (in-order)
      } else {
        __builtin_amdgcn_s_wait_tensorcnt(0);
      }
    }
    __syncthreads();   // publish buffer `buf` to all waves

    v16h afrag;
    {
      const f16* base = &ldsA[buf][(wm * 16 + r) * LSTR];
      uint32_t* a32 = (uint32_t*)&afrag;
      #pragma unroll
      for (int v = 0; v < 8; ++v) a32[v] = *(const uint32_t*)&base[kbaseA(v, hs)];
    }
    #pragma unroll
    for (int nt = 0; nt < 2; ++nt) {
      v16h bfrag;
      const f16* base = &ldsB[buf][(wn * 32 + nt * 16 + r) * LSTR];
      uint32_t* b32 = (uint32_t*)&bfrag;
      #pragma unroll
      for (int v = 0; v < 8; ++v) b32[v] = *(const uint32_t*)&base[kbaseB(v, hs)];
      acc[nt] = __builtin_amdgcn_wmma_f32_16x16x32_f16(
          false, afrag, false, bfrag, (short)0, acc[nt], false, false);
    }
    __syncthreads();   // all reads of `buf` done before TDM refills it (kt+2)
  }

  #pragma unroll
  for (int nt = 0; nt < 2; ++nt) {
    const int gcol = bn0 + wn * 32 + nt * 16 + r;
    #pragma unroll
    for (int i = 0; i < 8; ++i) {
      const int grow = bm0 + wm * 16 + hs * 8 + i;
      float vv = acc[nt][i];
      if (EPI == 1) vv = vv / (1.f + __expf(-1.702f * vv));  // QuickGELU
      if (EPI == 2)
        outF[(size_t)grow * N + gcol] = vv + resid[(size_t)grow * N + gcol];
      else
        outH[(size_t)grow * N + gcol] = (f16)vv;
    }
  }
}

// ---------------------------------------------------------------------------
// Flash attention. qkv f16 [token, 2304] (q|k|v). Block = (64 q rows, head),
// 128 threads = 4 waves x 16 q-rows. 32-key tiles: K tile fetched by TDM
// (overlapped with the manual transposed-V staging), S and P*V via WMMA,
// online softmax in C-layout registers (+__shfl_xor row reductions).
// ---------------------------------------------------------------------------
#define KSTR 68   // ldsK row stride (64 dims + 4; TDM pad 32dw/2dw)
#define VSTR 36   // ldsVt row stride (32 keys + 4)
#define PSTR 36   // ldsP  row stride (32 keys + 4)

__global__ __launch_bounds__(128) void attn_kernel(
    const f16* __restrict__ qkv, f16* __restrict__ z) {
  __shared__ f16 ldsK[32 * KSTR];       // [key][dim]   (TDM-filled)
  __shared__ f16 ldsVt[HDIM * VSTR];    // [dim][key]   (transposed for B-frags)
  __shared__ f16 ldsP[4 * 16 * PSTR];   // per-wave P tile [m][key]
  const int tid = threadIdx.x;
  const int lane = tid & 31, wave = tid >> 5;
  const int r = lane & 15, hs = lane >> 4;
  const int bh = blockIdx.y;
  const int b = bh / NHEAD, h = bh % NHEAD;
  const int q0 = blockIdx.x * 64 + wave * 16;
  const int c0 = h * HDIM;

  // Q fragments: rows q0..q0+15, K-dims 0..63 (two k-steps of 32)
  v16h qf[2];
  {
    const int t = (q0 + r) * BATCH + b;
    const f16* qrow = qkv + (size_t)t * D_QKV + c0;
    #pragma unroll
    for (int ks = 0; ks < 2; ++ks) {
      uint32_t* a32 = (uint32_t*)&qf[ks];
      #pragma unroll
      for (int v = 0; v < 8; ++v)
        a32[v] = *(const uint32_t*)&qrow[ks * 32 + kbaseA(v, hs)];
    }
  }

  v8f o[4] = {};
  float mrow[8], lrow[8];
  #pragma unroll
  for (int i = 0; i < 8; ++i) { mrow[i] = -1e30f; lrow[i] = 0.f; }

  for (int kt = 0; kt < SEQ_L / 32; ++kt) {
    __syncthreads();   // previous iteration's LDS reads complete
    // K tile (32 keys x 64 dims, row stride B*3D) via TDM, wave 0
    if (wave == 0) {
      const int t0 = (kt * 32) * BATCH + b;
      tdm_load_2d_f16(lds_off(&ldsK[0]),
                      (uint64_t)(uintptr_t)(qkv + (size_t)t0 * D_QKV + D_MODEL + c0),
                      HDIM, 32, BATCH * D_QKV, HDIM, 32,
                      /*pad 32 dwords*/ 4, /*by 2 dwords*/ 1);
    }
    // V tile staged transposed by all waves (overlaps the TDM)
    #pragma unroll
    for (int j = 0; j < 8; ++j) {
      const int idx = j * 128 + tid;            // 0..1023
      const int key = idx >> 5, dp = idx & 31;  // dim pair
      const int t = (kt * 32 + key) * BATCH + b;
      union { uint32_t u; f16 x[2]; } cv;
      cv.u = *(const uint32_t*)(qkv + (size_t)t * D_QKV + 2 * D_MODEL + c0 + dp * 2);
      ldsVt[(dp * 2) * VSTR + key]     = cv.x[0];
      ldsVt[(dp * 2 + 1) * VSTR + key] = cv.x[1];
    }
    if (wave == 0) __builtin_amdgcn_s_wait_tensorcnt(0);
    __syncthreads();   // K (TDM) + Vt visible to all waves

    // S = Q @ K^T  (16 x 32): two key n-tiles, two HDIM k-steps each
    v8f s[2];
    #pragma unroll
    for (int nt = 0; nt < 2; ++nt) {
      v8f c = {};
      #pragma unroll
      for (int ks = 0; ks < 2; ++ks) {
        v16h kf;
        const f16* base = &ldsK[(nt * 16 + r) * KSTR + ks * 32];
        uint32_t* b32 = (uint32_t*)&kf;
        #pragma unroll
        for (int v = 0; v < 8; ++v) b32[v] = *(const uint32_t*)&base[kbaseB(v, hs)];
        c = __builtin_amdgcn_wmma_f32_16x16x32_f16(
            false, qf[ks], false, kf, (short)0, c, false, false);
      }
      s[nt] = c;
    }

    // online softmax (1/sqrt(64) folded here)
    float p0[8], p1[8], scale[8];
    #pragma unroll
    for (int i = 0; i < 8; ++i) {
      const float a0 = s[0][i] * 0.125f, a1 = s[1][i] * 0.125f;
      float mx = fmaxf(a0, a1);
      #pragma unroll
      for (int msk = 8; msk >= 1; msk >>= 1) mx = fmaxf(mx, __shfl_xor(mx, msk, 32));
      const float mnew = fmaxf(mrow[i], mx);
      p0[i] = __expf(a0 - mnew);
      p1[i] = __expf(a1 - mnew);
      float rsum = p0[i] + p1[i];
      #pragma unroll
      for (int msk = 8; msk >= 1; msk >>= 1) rsum += __shfl_xor(rsum, msk, 32);
      scale[i] = __expf(mrow[i] - mnew);
      lrow[i] = lrow[i] * scale[i] + rsum;
      mrow[i] = mnew;
    }
    #pragma unroll
    for (int dt = 0; dt < 4; ++dt)
      #pragma unroll
      for (int i = 0; i < 8; ++i) o[dt][i] *= scale[i];

    // re-swizzle P through LDS: C-layout -> A-fragment layout (same-wave LDS
    // ops are in-order)
    f16* pb = &ldsP[wave * 16 * PSTR];
    #pragma unroll
    for (int i = 0; i < 8; ++i) {
      pb[(hs * 8 + i) * PSTR + r]      = (f16)p0[i];
      pb[(hs * 8 + i) * PSTR + 16 + r] = (f16)p1[i];
    }
    v16h pf;
    {
      const f16* base = &pb[r * PSTR];
      uint32_t* a32 = (uint32_t*)&pf;
      #pragma unroll
      for (int v = 0; v < 8; ++v) a32[v] = *(const uint32_t*)&base[kbaseA(v, hs)];
    }
    // O += P @ V : four 16-dim tiles
    #pragma unroll
    for (int dt = 0; dt < 4; ++dt) {
      v16h vf;
      const f16* base = &ldsVt[(dt * 16 + r) * VSTR];
      uint32_t* b32 = (uint32_t*)&vf;
      #pragma unroll
      for (int v = 0; v < 8; ++v) b32[v] = *(const uint32_t*)&base[kbaseB(v, hs)];
      o[dt] = __builtin_amdgcn_wmma_f32_16x16x32_f16(
          false, pf, false, vf, (short)0, o[dt], false, false);
    }
  }

  // normalize and write z[token, D] as f16 (out-proj input layout)
  #pragma unroll
  for (int dt = 0; dt < 4; ++dt) {
    #pragma unroll
    for (int i = 0; i < 8; ++i) {
      const int qrow = q0 + hs * 8 + i;
      const int t = qrow * BATCH + b;
      z[(size_t)t * D_MODEL + c0 + dt * 16 + r] = (f16)(o[dt][i] / lrow[i]);
    }
  }
}

// ---------------------------------------------------------------------------
// Host-side orchestration
// ---------------------------------------------------------------------------
static inline size_t alignup(size_t x) { return (x + 255) & ~(size_t)255; }

extern "C" void kernel_launch(void* const* d_in, const int* in_sizes, int n_in,
                              void* d_out, int out_size, void* d_ws, size_t ws_size,
                              hipStream_t stream) {
  (void)in_sizes; (void)n_in; (void)out_size; (void)ws_size;
  const float* x     = (const float*)d_in[0];
  const float* w_in  = (const float*)d_in[1];
  const float* b_in  = (const float*)d_in[2];
  const float* w_out = (const float*)d_in[3];
  const float* b_out = (const float*)d_in[4];
  const float* g1    = (const float*)d_in[5];
  const float* be1   = (const float*)d_in[6];
  const float* g2    = (const float*)d_in[7];
  const float* be2   = (const float*)d_in[8];
  const float* w1    = (const float*)d_in[9];
  const float* b1    = (const float*)d_in[10];
  const float* w2    = (const float*)d_in[11];
  const float* b2    = (const float*)d_in[12];
  float* out = (float*)d_out;

  char* ws = (char*)d_ws;
  size_t off = 0;
  f16* wInH  = (f16*)(ws + off); off += alignup((size_t)D_QKV * D_MODEL * 2);
  f16* wOutH = (f16*)(ws + off); off += alignup((size_t)D_MODEL * D_MODEL * 2);
  f16* w1H   = (f16*)(ws + off); off += alignup((size_t)D_FF * D_MODEL * 2);
  f16* w2H   = (f16*)(ws + off); off += alignup((size_t)D_MODEL * D_FF * 2);
  f16* h1    = (f16*)(ws + off); off += alignup((size_t)N_TOK * D_MODEL * 2);
  f16* qkv   = (f16*)(ws + off); off += alignup((size_t)N_TOK * D_QKV * 2);
  f16* zbuf  = (f16*)(ws + off); off += alignup((size_t)N_TOK * D_MODEL * 2);
  float* x2  = (float*)(ws + off); off += alignup((size_t)N_TOK * D_MODEL * 4);
  f16* h2    = (f16*)(ws + off); off += alignup((size_t)N_TOK * D_MODEL * 2);
  f16* a1    = (f16*)(ws + off); off += alignup((size_t)N_TOK * D_FF * 2);

  // 0) weight downcast
  {
    int n;
    n = D_QKV * D_MODEL;
    f32_to_f16_kernel<<<(n + 255) / 256, 256, 0, stream>>>(w_in, wInH, n);
    n = D_MODEL * D_MODEL;
    f32_to_f16_kernel<<<(n + 255) / 256, 256, 0, stream>>>(w_out, wOutH, n);
    n = D_FF * D_MODEL;
    f32_to_f16_kernel<<<(n + 255) / 256, 256, 0, stream>>>(w1, w1H, n);
    n = D_MODEL * D_FF;
    f32_to_f16_kernel<<<(n + 255) / 256, 256, 0, stream>>>(w2, w2H, n);
  }

  // 1) h1 = LN1(x)
  layernorm_kernel<<<N_TOK, 256, 0, stream>>>(x, g1, be1, h1);

  // 2) qkv = h1 @ w_in^T + b_in
  gemm_kernel<0><<<dim3(N_TOK / BM, D_QKV / BN), 256, 0, stream>>>(
      h1, wInH, b_in, N_TOK, D_QKV, D_MODEL, qkv, nullptr, nullptr);

  // 3) flash attention -> z
  attn_kernel<<<dim3(SEQ_L / 64, BATCH * NHEAD), 128, 0, stream>>>(qkv, zbuf);

  // 4) x2 = x + z @ w_out^T + b_out
  gemm_kernel<2><<<dim3(N_TOK / BM, D_MODEL / BN), 256, 0, stream>>>(
      zbuf, wOutH, b_out, N_TOK, D_MODEL, D_MODEL, nullptr, x2, x);

  // 5) h2 = LN2(x2)
  layernorm_kernel<<<N_TOK, 256, 0, stream>>>(x2, g2, be2, h2);

  // 6) a1 = QuickGELU(h2 @ w1^T + b1)
  gemm_kernel<1><<<dim3(N_TOK / BM, D_FF / BN), 256, 0, stream>>>(
      h2, w1H, b1, N_TOK, D_FF, D_MODEL, a1, nullptr, nullptr);

  // 7) out = x2 + a1 @ w2^T + b2
  gemm_kernel<2><<<dim3(N_TOK / BM, D_MODEL / BN), 256, 0, stream>>>(
      a1, w2H, b2, N_TOK, D_MODEL, D_FF, nullptr, out, x2);
}
